// CharDecoder_72258529788356
// MI455X (gfx1250) — matrix operand order repaired and checked by reading, not compiled, not run on hardware
//
#include <hip/hip_runtime.h>
#include <hip/hip_bf16.h>
#include <stdint.h>

#define Lt 32
#define Bk 16384
#define Hd 256
#define Ed 50
#define Vv 96

typedef __attribute__((ext_vector_type(16))) __bf16 v16bf;
typedef __attribute__((ext_vector_type(8)))  float  v8f;

union Frag { v16bf v; uint4 q[2]; };

static __device__ __forceinline__ v8f splat8(float x) {
    v8f r;
#pragma unroll
    for (int i = 0; i < 8; ++i) r[i] = x;
    return r;
}

static __device__ __forceinline__ unsigned short f2bf(float f) {
    unsigned u = __builtin_bit_cast(unsigned, f);
    unsigned r = (u + 0x7FFFu + ((u >> 16) & 1u)) >> 16;   // round-to-nearest-even
    return (unsigned short)r;
}

static __device__ __forceinline__ v16bf ldg_frag(const char* p) {
    Frag f;
    f.q[0] = *(const uint4*)p;
    f.q[1] = *(const uint4*)(p + 16);
    return f.v;
}

// ---------------------------------------------------------------------------
// bsum = b_ih + b_hh
__global__ void kbias(const float* __restrict__ bih, const float* __restrict__ bhh,
                      float* __restrict__ bsum) {
    int i = blockIdx.x * blockDim.x + threadIdx.x;
    if (i < 4 * Hd) bsum[i] = bih[i] + bhh[i];
}

// ---------------------------------------------------------------------------
// Pack a row-major fp32 weight W[n][k] (n = GEMM output col, k = reduction)
// into bf16 WMMA B-fragments: frag(ntile, kt) -> lane = n (l & 15) within 16-tile,
// halves = K 8*(l>>4)+0..7, 16+8*(l>>4)+0..7.  blockIdx.x = ntile*ktiles + kt, blockDim = 32.
__global__ void kpackw(const float* __restrict__ src, char* __restrict__ dst,
                       int ktiles, int kreal, int kpitch) {
    int l = threadIdx.x;                 // n = l & 15, k half = l >> 4
    int ntile = blockIdx.x / ktiles;
    int kt = blockIdx.x % ktiles;
    int n = ntile * 16 + (l & 15);
    union { unsigned short h[16]; uint4 q[2]; } u;
#pragma unroll
    for (int s = 0; s < 16; ++s) {
        int k = kt * 32 + (l >> 4) * 8 + (s < 8 ? s : s + 8);
        float v = (k < kreal) ? src[(size_t)n * kpitch + k] : 0.0f;
        u.h[s] = f2bf(v);
    }
    uint4* o = (uint4*)(dst + (size_t)blockIdx.x * 1024 + (size_t)l * 32);
    o[0] = u.q[0];
    o[1] = u.q[1];
}

// ---------------------------------------------------------------------------
// Gather embeddings for all (t, b) into bf16 WMMA A-fragments, K padded 50->64.
// A layout (16x32 bf16): lane m (0..15) row m, halves = K 0..7,16..23;
// lane m+16 row m, halves = K 8..15,24..31.
__global__ void kpackx(const int* __restrict__ tok, const float* __restrict__ emb,
                       char* __restrict__ dst) {
    unsigned gid = blockIdx.x * 256u + threadIdx.x;
    int lane  = gid & 31;
    int kt    = (gid >> 5) & 1;
    int mtile = (gid >> 6) & 1023;
    int t     = gid >> 16;
    int lo = lane & 15, hi = (lane >> 4) & 1;
    int b = mtile * 16 + lo;
    int tk = tok[(size_t)t * Bk + b];
    union { unsigned short h[16]; uint4 q[2]; } u;
#pragma unroll
    for (int hh = 0; hh < 16; ++hh) {
        int k = kt * 32 + hi * 8 + (hh < 8 ? hh : hh + 8);
        float v = (k < Ed) ? emb[(size_t)tk * Ed + k] : 0.0f;
        u.h[hh] = f2bf(v);
    }
    uint4* o = (uint4*)(dst + ((size_t)(t * 1024 + mtile) * 2 + kt) * 1024 + (size_t)lane * 32);
    o[0] = u.q[0];
    o[1] = u.q[1];
}

// ---------------------------------------------------------------------------
// Persistent per-workgroup LSTM: 64 batch rows / WG, 16 waves, wave j owns
// hidden columns 16j..16j+15. c carried in VGPRs, h shared via LDS (bf16).
__global__ __launch_bounds__(512) void lstm_main(
    const char* __restrict__ xfrag, const char* __restrict__ wih,
    const char* __restrict__ whh,   const char* __restrict__ wprj,
    const float* __restrict__ bsum, const float* __restrict__ bproj,
    const float* __restrict__ h0,   const float* __restrict__ c0,
    float* __restrict__ out)
{
    __shared__ unsigned short hbuf[64][264];   // pitch 264 halves -> bank-friendly

    const int w    = threadIdx.x >> 5;   // wave id == hidden tile j
    const int lane = threadIdx.x & 31;
    const int lo   = lane & 15;
    const int hi   = lane >> 4;
    const int j    = w;
    const int row0 = blockIdx.x * 64;

    float biasg[4];
#pragma unroll
    for (int g = 0; g < 4; ++g) biasg[g] = bsum[g * Hd + j * 16 + lo];

    // c state in registers: C-layout tiles (m, j)
    v8f cst[4];
#pragma unroll
    for (int m = 0; m < 4; ++m) {
#pragma unroll
        for (int r = 0; r < 8; ++r) {
            int row = row0 + m * 16 + r + 8 * hi;
            cst[m][r] = c0[(size_t)row * Hd + j * 16 + lo];
        }
    }

    // h0 -> LDS (bf16)
    for (int e = threadIdx.x; e < 64 * Hd; e += 512) {
        int r = e >> 8, col = e & 255;
        hbuf[r][col] = f2bf(h0[(size_t)(row0 + r) * Hd + col]);
    }
    __syncthreads();

    const size_t state_off = (size_t)Lt * Bk * Vv;

    for (int t = 0; t < Lt; ++t) {
        // prefetch next step's X fragments into cache
        if (t + 1 < Lt) {
#pragma unroll
            for (int m = 0; m < 4; ++m)
                __builtin_prefetch(xfrag + ((size_t)((t + 1) * 1024 + blockIdx.x * 4 + m) * 2) * 1024
                                         + (size_t)lane * 32, 0, 3);
        }

        v8f acc[4][4];
#pragma unroll
        for (int m = 0; m < 4; ++m)
#pragma unroll
            for (int g = 0; g < 4; ++g) acc[m][g] = splat8(biasg[g]);

        // -------- x_t @ W_ih^T  (K = 64 padded) --------
#pragma unroll 1
        for (int kt = 0; kt < 2; ++kt) {
            const char* wbase = wih + ((size_t)j * 2 + kt) * 1024 + (size_t)lane * 32;
            v16bf a[4];
#pragma unroll
            for (int m = 0; m < 4; ++m)
                a[m] = ldg_frag(xfrag + ((size_t)(t * 1024 + blockIdx.x * 4 + m) * 2 + kt) * 1024
                                      + (size_t)lane * 32);
            v16bf bb = ldg_frag(wbase);               // gate 0
#pragma unroll
            for (int g = 0; g < 4; ++g) {
                // issue next gate's B load BEFORE consuming current one
                v16bf bnext = (g < 3) ? ldg_frag(wbase + (size_t)(g + 1) * (16 * 2 * 1024)) : bb;
#pragma unroll
                for (int m = 0; m < 4; ++m)
                    acc[m][g] = __builtin_amdgcn_wmma_f32_16x16x32_bf16(
                        false, a[m], false, bb, (short)0, acc[m][g], false, false);
                bb = bnext;
            }
        }

        // -------- h_{t-1} @ W_hh^T  (K = 256) --------
#pragma unroll 1
        for (int kt = 0; kt < 8; ++kt) {
            const int k0 = kt * 32 + hi * 8;
            const char* wbase = whh + ((size_t)j * 8 + kt) * 1024 + (size_t)lane * 32;
            v16bf a[4];
#pragma unroll
            for (int m = 0; m < 4; ++m) {
                Frag f;
                const unsigned short* rp = &hbuf[m * 16 + lo][0];
                f.q[0] = *(const uint4*)(rp + k0);
                f.q[1] = *(const uint4*)(rp + k0 + 16);
                a[m] = f.v;
            }
            v16bf bb = ldg_frag(wbase);               // gate 0
#pragma unroll
            for (int g = 0; g < 4; ++g) {
                v16bf bnext = (g < 3) ? ldg_frag(wbase + (size_t)(g + 1) * (16 * 8 * 1024)) : bb;
#pragma unroll
                for (int m = 0; m < 4; ++m)
                    acc[m][g] = __builtin_amdgcn_wmma_f32_16x16x32_bf16(
                        false, a[m], false, bb, (short)0, acc[m][g], false, false);
                bb = bnext;
            }
        }

        __syncthreads();   // all reads of h(t-1) complete

        // -------- elementwise LSTM cell, write h_t to LDS --------
#pragma unroll
        for (int m = 0; m < 4; ++m) {
#pragma unroll
            for (int r = 0; r < 8; ++r) {
                float xi = acc[m][0][r], xf = acc[m][1][r];
                float xg = acc[m][2][r], xo = acc[m][3][r];
                float si = 1.0f / (1.0f + __expf(-xi));
                float sf = 1.0f / (1.0f + __expf(-xf));
                float tg = tanhf(xg);
                float so = 1.0f / (1.0f + __expf(-xo));
                float cn = sf * cst[m][r] + si * tg;
                cst[m][r] = cn;
                float hn = so * tanhf(cn);
                hbuf[m * 16 + r + 8 * hi][j * 16 + lo] = f2bf(hn);
                if (t == Lt - 1) {
                    int row = row0 + m * 16 + r + 8 * hi;
                    out[state_off + (size_t)row * Hd + j * 16 + lo] = hn;                      // h_T
                    out[state_off + (size_t)Bk * Hd + (size_t)row * Hd + j * 16 + lo] = cn;    // c_T
                }
            }
        }

        __syncthreads();   // h_t visible to all waves

        // -------- scores_t = h_t @ W_proj^T + b_proj  (24 tiles over 16 waves) --------
        for (int tile = w; tile < 24; tile += 16) {
            int m = tile / 6, vt = tile % 6;
            v8f pacc = splat8(bproj[vt * 16 + lo]);
            const char* wbase = wprj + (size_t)vt * 8 * 1024 + (size_t)lane * 32;
            v16bf bb = ldg_frag(wbase);               // kt = 0
#pragma unroll 1
            for (int kt = 0; kt < 8; ++kt) {
                const int k0 = kt * 32 + hi * 8;
                Frag f;
                const unsigned short* rp = &hbuf[m * 16 + lo][0];
                f.q[0] = *(const uint4*)(rp + k0);
                f.q[1] = *(const uint4*)(rp + k0 + 16);
                v16bf bnext = (kt < 7) ? ldg_frag(wbase + (size_t)(kt + 1) * 1024) : bb;
                pacc = __builtin_amdgcn_wmma_f32_16x16x32_bf16(
                    false, f.v, false, bb, (short)0, pacc, false, false);
                bb = bnext;
            }
#pragma unroll
            for (int r = 0; r < 8; ++r) {
                int row = row0 + m * 16 + r + 8 * hi;
                out[(size_t)t * Bk * Vv + (size_t)row * Vv + vt * 16 + lo] = pacc[r];
            }
        }
        // next iteration's gate reads of hbuf are ordered by its own barrier
    }
}

// ---------------------------------------------------------------------------
extern "C" void kernel_launch(void* const* d_in, const int* in_sizes, int n_in,
                              void* d_out, int out_size, void* d_ws, size_t ws_size,
                              hipStream_t stream) {
    const int*   tok  = (const int*)  d_in[0];
    const float* h0   = (const float*)d_in[1];
    const float* c0   = (const float*)d_in[2];
    const float* emb  = (const float*)d_in[3];
    const float* Wih  = (const float*)d_in[4];
    const float* Whh  = (const float*)d_in[5];
    const float* bih  = (const float*)d_in[6];
    const float* bhh  = (const float*)d_in[7];
    const float* Wprj = (const float*)d_in[8];
    const float* bprj = (const float*)d_in[9];
    float* out = (float*)d_out;

    char* ws = (char*)d_ws;
    const size_t XOFF = 0;
    const size_t XSZ  = (size_t)Lt * 1024 * 2 * 1024;   // 64 MB of X fragments
    const size_t WIHO = XOFF + XSZ,  WIHS = (size_t)64 * 2 * 1024;
    const size_t WHHO = WIHO + WIHS, WHHS = (size_t)64 * 8 * 1024;
    const size_t WPO  = WHHO + WHHS, WPS  = (size_t)6 * 8 * 1024;
    const size_t BSO  = WPO + WPS;

    kbias <<<4, 256, 0, stream>>>(bih, bhh, (float*)(ws + BSO));
    kpackw<<<64 * 2, 32, 0, stream>>>(Wih,  ws + WIHO, 2, Ed, Ed);   // K padded 50->64
    kpackw<<<64 * 8, 32, 0, stream>>>(Whh,  ws + WHHO, 8, Hd, Hd);
    kpackw<<<6 * 8,  32, 0, stream>>>(Wprj, ws + WPO,  8, Hd, Hd);
    kpackx<<<8192, 256, 0, stream>>>(tok, emb, ws + XOFF);

    lstm_main<<<Bk / 64, 512, 0, stream>>>(ws + XOFF, ws + WIHO, ws + WHHO, ws + WPO,
                                           (const float*)(ws + BSO), bprj, h0, c0, out);
}
